// FEN_46488726011915
// MI455X (gfx1250) — matrix operand order repaired and checked by reading, not compiled, not run on hardware
//
#include <hip/hip_runtime.h>

// ---------------- model constants ----------------
#define N_EMBD 256
#define N_INPUTS 66
#define N_LEVELS 8
#define AND_PER_LEVEL 4096
#define NOT_PER_LEVEL 2048
#define LEVEL (AND_PER_LEVEL + NOT_PER_LEVEL)
#define N_NODES (N_INPUTS + N_LEVELS * LEVEL)   // 49218
#define N_HIDDEN 1024
#define N_HID_LAYERS 8

typedef __attribute__((ext_vector_type(16))) __bf16 v16bf;
typedef __attribute__((ext_vector_type(8)))  float  v8f;

// ---- gfx1250 async global->LDS path (ASYNCcnt), with safe fallback ----
#if defined(__gfx1250__) && defined(__has_builtin)
#if __has_builtin(__builtin_amdgcn_global_load_async_to_lds_b128)
#define USE_ASYNC_LDS 1
#endif
#endif
#ifndef USE_ASYNC_LDS
#define USE_ASYNC_LDS 0
#endif

#if USE_ASYNC_LDS
#define GLOBAL_AS __attribute__((address_space(1)))
#define LDS_AS    __attribute__((address_space(3)))
// builtin signature (from clang diagnostic): (int4 AS1*, int4 AS3*, imm int, imm int)
// gcc-style vector to match the builtin's pointee type exactly
typedef int v4i_b __attribute__((vector_size(16)));
// generic->AS1 is identity; generic LDS pointer's low 32 bits are the LDS offset
#define ASYNC_COPY16(dst_lds, src_glb)                                          \
    __builtin_amdgcn_global_load_async_to_lds_b128(                             \
        (GLOBAL_AS v4i_b*)(unsigned long long)(src_glb),                        \
        (LDS_AS v4i_b*)(unsigned int)(unsigned long long)(dst_lds), 0, 0)
#if __has_builtin(__builtin_amdgcn_s_wait_asynccnt)
#define WAIT_ASYNC() __builtin_amdgcn_s_wait_asynccnt(0)
#else
#define WAIT_ASYNC() asm volatile("s_wait_asynccnt 0x0" ::: "memory")
#endif
#endif

// fp32 -> bf16 round-to-nearest-even (bit trick)
__device__ __forceinline__ unsigned short f32_to_bf16(float f) {
    unsigned int u = __float_as_uint(f);
    unsigned int r = (u + 0x7FFFu + ((u >> 16) & 1u)) >> 16;
    return (unsigned short)r;
}

// ---------------- weight prep: W[K][N] fp32 -> Wt[N][K] bf16 ----------------
__global__ void transpose_to_bf16(const float* __restrict__ W,
                                  unsigned short* __restrict__ Wt,
                                  int K, int N) {
    long idx = (long)blockIdx.x * blockDim.x + threadIdx.x;
    long total = (long)K * N;
    if (idx >= total) return;
    int k = (int)(idx % K);
    int n = (int)(idx / K);
    Wt[idx] = f32_to_bf16(W[(long)k * N + n]);
}

// ---------------- node table init (inputs region) ----------------
__global__ void init_inputs(const float* __restrict__ embd,
                            float* __restrict__ node,
                            unsigned short* __restrict__ nodeb, int count) {
    int i = blockIdx.x * blockDim.x + threadIdx.x;
    if (i >= count) return;
    float v = embd[i];
    node[i] = v;
    nodeb[i] = f32_to_bf16(v);
}

// ---------------- NOT nodes: dst = -node[src] ----------------
__global__ void not_kernel(const int* __restrict__ x_edges,
                           float* __restrict__ node,
                           unsigned short* __restrict__ nodeb, int s) {
    int t = blockIdx.x * blockDim.x + threadIdx.x;   // NOT_PER_LEVEL * 256
    int i = t >> 8, c = t & 255;
    int dst = s + AND_PER_LEVEL + i;
    int src = x_edges[dst];
    float v = -node[(size_t)src * N_EMBD + c];
    node[(size_t)dst * N_EMBD + c]  = v;
    nodeb[(size_t)dst * N_EMBD + c] = f32_to_bf16(v);
}

// ---------------- AND gather: h0[4096][512] = concat(node[xa], node[ya]) (bf16) --------
__global__ void gather_kernel(const int* __restrict__ x_edges,
                              const int* __restrict__ y_edges,
                              const unsigned short* __restrict__ nodeb,
                              unsigned short* __restrict__ h0, int s) {
    int t = blockIdx.x * blockDim.x + threadIdx.x;   // 4096 * 512
    int row = t >> 9, col = t & 511;
    int src = (col < N_EMBD) ? x_edges[s + row] : y_edges[s + row];
    h0[t] = nodeb[(size_t)src * N_EMBD + (col & (N_EMBD - 1))];
}

// ---------------- WMMA GEMM: C[M][N] = A[M][K](bf16) x Bt[N][K](bf16)^T + bias -----------
// block tile 128x128, 256 threads = 8 waves in 2(M) x 4(N); wave tile 64x32 = 8 WMMA accs.
// Double-buffered LDS; steady state = 1 barrier per 64-wide k-block; next k-block's
// global traffic issued (async-to-LDS when available) before the current 128 WMMAs.
__global__ __launch_bounds__(256) void gemm_bf16_wmma(
    const unsigned short* __restrict__ A,    // [M][K] bf16 row-major
    const unsigned short* __restrict__ Bt,   // [N][K] bf16 (weights pre-transposed)
    const float* __restrict__ bias,          // [N]
    unsigned short* __restrict__ Obf,        // [M][N] bf16 out (or null)
    float* __restrict__ Of32,                // [M][N] f32 out (or null)
    int K, int N, int relu) {
    __shared__ __align__(16) unsigned short sA[2][128][72];  // +8 pad: conflict-free b128 reads
    __shared__ __align__(16) unsigned short sB[2][128][72];

    const int tid    = threadIdx.x;
    const int lane   = tid & 31;
    const int wave   = tid >> 5;
    const int laneLo = lane & 15;
    const int laneHi = lane >> 4;
    const int waveM  = wave & 1;     // 0..1
    const int waveN  = wave >> 1;    // 0..3
    const int blockN = blockIdx.x;
    const int blockM = blockIdx.y;

    v8f acc[4][2];
    const v8f vzero = {0.f, 0.f, 0.f, 0.f, 0.f, 0.f, 0.f, 0.f};
#pragma unroll
    for (int i = 0; i < 4; ++i)
#pragma unroll
        for (int j = 0; j < 2; ++j) acc[i][j] = vzero;

    const int lrow = tid >> 3;        // 0..31
    const int lcol = (tid & 7) << 3;  // 0,8,..,56
    const size_t aBase = (size_t)blockM * 128 * K;
    const size_t bBase = (size_t)blockN * 128 * K;
    const int nBlk = K >> 6;

    // ---- prologue: fill buffer 0 with k-block 0 ----
#if USE_ASYNC_LDS
#pragma unroll
    for (int p = 0; p < 4; ++p) {
        int row = p * 32 + lrow;
        ASYNC_COPY16(&sA[0][row][lcol], A  + aBase + (size_t)row * K + lcol);
        ASYNC_COPY16(&sB[0][row][lcol], Bt + bBase + (size_t)row * K + lcol);
    }
    WAIT_ASYNC();
    __syncthreads();
#else
    uint4 ra[4], rb[4];
#pragma unroll
    for (int p = 0; p < 4; ++p) {
        int row = p * 32 + lrow;
        ra[p] = *reinterpret_cast<const uint4*>(A  + aBase + (size_t)row * K + lcol);
        rb[p] = *reinterpret_cast<const uint4*>(Bt + bBase + (size_t)row * K + lcol);
    }
#pragma unroll
    for (int p = 0; p < 4; ++p) {
        int row = p * 32 + lrow;
        *reinterpret_cast<uint4*>(&sA[0][row][lcol]) = ra[p];
        *reinterpret_cast<uint4*>(&sB[0][row][lcol]) = rb[p];
    }
    __syncthreads();
#endif

    for (int kb = 0; kb < nBlk; ++kb) {
        const int  cur  = kb & 1;
        const int  nxt  = cur ^ 1;
        const bool more = (kb + 1) < nBlk;
        const int  kOfs = (kb + 1) << 6;

        // ---- issue next k-block's global traffic before computing ----
#if USE_ASYNC_LDS
        if (more) {
#pragma unroll
            for (int p = 0; p < 4; ++p) {
                int row = p * 32 + lrow;
                ASYNC_COPY16(&sA[nxt][row][lcol], A  + aBase + (size_t)row * K + kOfs + lcol);
                ASYNC_COPY16(&sB[nxt][row][lcol], Bt + bBase + (size_t)row * K + kOfs + lcol);
            }
        }
#else
        if (more) {
#pragma unroll
            for (int p = 0; p < 4; ++p) {
                int row = p * 32 + lrow;
                ra[p] = *reinterpret_cast<const uint4*>(A  + aBase + (size_t)row * K + kOfs + lcol);
                rb[p] = *reinterpret_cast<const uint4*>(Bt + bBase + (size_t)row * K + kOfs + lcol);
            }
        }
#endif

        // ---- compute on buffer `cur`: 2 ksteps x 8 WMMA ----
#pragma unroll
        for (int ks = 0; ks < 2; ++ks) {
            const int ko = ks * 32;
            // B fragments: lane holds column n = laneLo, K = 16*laneHi .. +15 (contiguous)
            union { uint4 u[2]; v16bf v; } bf[2];
#pragma unroll
            for (int j = 0; j < 2; ++j) {
                int n = waveN * 32 + j * 16 + laneLo;
                const uint4* p0 = reinterpret_cast<const uint4*>(&sB[cur][n][ko + laneHi * 16]);
                bf[j].u[0] = p0[0];
                bf[j].u[1] = p0[1];
            }
            // A fragments: lane holds row m = laneLo, K = 16*laneHi .. +15 (contiguous)
#pragma unroll
            for (int i = 0; i < 4; ++i) {
                int m = waveM * 64 + i * 16 + laneLo;
                union { uint4 u[2]; v16bf v; } af;
                af.u[0] = *reinterpret_cast<const uint4*>(&sA[cur][m][ko + laneHi * 16]);
                af.u[1] = *reinterpret_cast<const uint4*>(&sA[cur][m][ko + laneHi * 16 + 8]);
#pragma unroll
                for (int j = 0; j < 2; ++j) {
                    acc[i][j] = __builtin_amdgcn_wmma_f32_16x16x32_bf16(
                        false, af.v, false, bf[j].v, (short)0, acc[i][j], false, false);
                }
            }
        }

        // ---- commit next buffer, single barrier per iteration ----
#if USE_ASYNC_LDS
        if (more) WAIT_ASYNC();
        __syncthreads();
#else
        if (more) {
#pragma unroll
            for (int p = 0; p < 4; ++p) {
                int row = p * 32 + lrow;
                *reinterpret_cast<uint4*>(&sA[nxt][row][lcol]) = ra[p];
                *reinterpret_cast<uint4*>(&sB[nxt][row][lcol]) = rb[p];
            }
        }
        __syncthreads();
#endif
    }

    // Epilogue: C layout — lane holds column n = laneLo; VGPR r holds row m = 8*laneHi + r
#pragma unroll
    for (int j = 0; j < 2; ++j) {
        int nn = blockN * 128 + waveN * 32 + j * 16 + laneLo;
        float bv = bias[nn];
#pragma unroll
        for (int i = 0; i < 4; ++i) {
            int mb = blockM * 128 + waveM * 64 + i * 16 + laneHi * 8;
#pragma unroll
            for (int r = 0; r < 8; ++r) {
                float v = acc[i][j][r] + bv;
                if (relu) v = fmaxf(v, 0.f);
                size_t o = (size_t)(mb + r) * N + nn;
                if (Obf)  Obf[o]  = f32_to_bf16(v);
                if (Of32) Of32[o] = v;
            }
        }
    }
}

// ---------------- LayerNorm + scatter into node table ----------------
__global__ void ln_scatter_kernel(const float* __restrict__ xin,   // [4096][256]
                                  const float* __restrict__ g,
                                  const float* __restrict__ b,
                                  float* __restrict__ node,
                                  unsigned short* __restrict__ nodeb, int s) {
    __shared__ float red[256];
    int row = blockIdx.x, c = threadIdx.x;
    float v = xin[(size_t)row * N_EMBD + c];
    red[c] = v;
    __syncthreads();
    for (int st = 128; st > 0; st >>= 1) {
        if (c < st) red[c] += red[c + st];
        __syncthreads();
    }
    float mu = red[0] * (1.0f / N_EMBD);
    __syncthreads();
    float d = v - mu;
    red[c] = d * d;
    __syncthreads();
    for (int st = 128; st > 0; st >>= 1) {
        if (c < st) red[c] += red[c + st];
        __syncthreads();
    }
    float var = red[0] * (1.0f / N_EMBD);
    float y = d * rsqrtf(var + 1e-5f) * g[c] + b[c];
    size_t o = (size_t)(s + row) * N_EMBD + c;
    node[o]  = y;
    nodeb[o] = f32_to_bf16(y);
}

// ---------------- host ----------------
extern "C" void kernel_launch(void* const* d_in, const int* in_sizes, int n_in,
                              void* d_out, int out_size, void* d_ws, size_t ws_size,
                              hipStream_t stream) {
    (void)in_sizes; (void)n_in; (void)out_size; (void)ws_size;
    const int*   x_edges    = (const int*)d_in[1];
    const int*   y_edges    = (const int*)d_in[2];
    const float* input_embd = (const float*)d_in[3];
    const float* W_in       = (const float*)d_in[4];
    const float* b_in       = (const float*)d_in[5];
    const float* W_hid      = (const float*)d_in[6];
    const float* b_hid      = (const float*)d_in[7];
    const float* W_out      = (const float*)d_in[8];
    const float* b_out      = (const float*)d_in[9];
    const float* ln_g       = (const float*)d_in[10];
    const float* ln_b       = (const float*)d_in[11];
    float* node = (float*)d_out;   // node_embd lives directly in d_out (fp32)

    // carve workspace (~69 MB total)
    char* ws = (char*)d_ws;
    size_t off = 0;
    auto carve = [&](size_t bytes) -> void* {
        void* p = ws + off;
        off += (bytes + 255) & ~(size_t)255;
        return p;
    };
    unsigned short* nodeb = (unsigned short*)carve((size_t)N_NODES * N_EMBD * 2); // bf16 mirror
    unsigned short* WtIn  = (unsigned short*)carve((size_t)N_HIDDEN * 512 * 2);
    unsigned short* WtHid = (unsigned short*)carve((size_t)N_HID_LAYERS * N_HIDDEN * N_HIDDEN * 2);
    unsigned short* WtOut = (unsigned short*)carve((size_t)N_EMBD * N_HIDDEN * 2);
    unsigned short* h0    = (unsigned short*)carve((size_t)AND_PER_LEVEL * 512 * 2);
    unsigned short* hA    = (unsigned short*)carve((size_t)AND_PER_LEVEL * N_HIDDEN * 2);
    unsigned short* hB    = (unsigned short*)carve((size_t)AND_PER_LEVEL * N_HIDDEN * 2);
    float*          outF  = (float*)carve((size_t)AND_PER_LEVEL * N_EMBD * 4);

    const int THR = 256;
    // one-time weight conversion + transpose to [N][K] bf16
    transpose_to_bf16<<<(512 * N_HIDDEN + THR - 1) / THR, THR, 0, stream>>>(W_in, WtIn, 512, N_HIDDEN);
    for (int i = 0; i < N_HID_LAYERS; ++i)
        transpose_to_bf16<<<(N_HIDDEN * N_HIDDEN + THR - 1) / THR, THR, 0, stream>>>(
            W_hid + (size_t)i * N_HIDDEN * N_HIDDEN, WtHid + (size_t)i * N_HIDDEN * N_HIDDEN,
            N_HIDDEN, N_HIDDEN);
    transpose_to_bf16<<<(N_HIDDEN * N_EMBD + THR - 1) / THR, THR, 0, stream>>>(W_out, WtOut, N_HIDDEN, N_EMBD);

    init_inputs<<<(N_INPUTS * N_EMBD + THR - 1) / THR, THR, 0, stream>>>(
        input_embd, node, nodeb, N_INPUTS * N_EMBD);

    for (int l = 0; l < N_LEVELS; ++l) {
        int s = N_INPUTS + l * LEVEL;
        not_kernel<<<NOT_PER_LEVEL * N_EMBD / THR, THR, 0, stream>>>(x_edges, node, nodeb, s);
        gather_kernel<<<AND_PER_LEVEL * 512 / THR, THR, 0, stream>>>(x_edges, y_edges, nodeb, h0, s);

        // input layer: [4096x512] @ [512x1024], ReLU
        gemm_bf16_wmma<<<dim3(N_HIDDEN / 128, AND_PER_LEVEL / 128), THR, 0, stream>>>(
            h0, WtIn, b_in, hA, nullptr, 512, N_HIDDEN, 1);

        unsigned short* cur = hA;
        unsigned short* nxt = hB;
        for (int i = 0; i < N_HID_LAYERS; ++i) {
            gemm_bf16_wmma<<<dim3(N_HIDDEN / 128, AND_PER_LEVEL / 128), THR, 0, stream>>>(
                cur, WtHid + (size_t)i * N_HIDDEN * N_HIDDEN, b_hid + (size_t)i * N_HIDDEN,
                nxt, nullptr, N_HIDDEN, N_HIDDEN, 1);
            unsigned short* t = cur; cur = nxt; nxt = t;
        }
        // output layer: [4096x1024] @ [1024x256], no ReLU, fp32 out for LN
        gemm_bf16_wmma<<<dim3(N_EMBD / 128, AND_PER_LEVEL / 128), THR, 0, stream>>>(
            cur, WtOut, b_out, nullptr, outF, N_HIDDEN, N_EMBD, 0);

        ln_scatter_kernel<<<AND_PER_LEVEL, N_EMBD, 0, stream>>>(outF, ln_g, ln_b, node, nodeb, s);
    }
}
